// DrugProteinModel_12661563588711
// MI455X (gfx1250) — matrix operand order repaired and checked by reading, not compiled, not run on hardware
//
#include <hip/hip_runtime.h>
#include <hip/hip_bf16.h>

// ---------------------------------------------------------------------------
// DrugProteinModel on gfx1250 (MI455X): edge mean-aggregate (f32 atomics,
// L2-resident), GEMMs via v_wmma_f32_16x16x32_f16 (f16 in / f32 acc),
// protein GEMM K-loop double-buffered with global_load_async_to_lds_b128.
// ---------------------------------------------------------------------------

#define NUM_NODES  50000
#define NUM_EDGES  800000
#define NUM_GRAPHS 1024
#define D_IN       64
#define D_PROT     1280
#define HID        256

typedef _Float16 half_t;
typedef __attribute__((ext_vector_type(16))) _Float16 v16h;
typedef __attribute__((ext_vector_type(8)))  float    v8f;

// ---- WMMA wrapper ----------------------------------------------------------
__device__ __forceinline__ v8f wmma_f16(v16h a, v16h b, v8f c) {
  // (neg_a, A, neg_b, B, c_mod, C, reuse_a, reuse_b)
  return __builtin_amdgcn_wmma_f32_16x16x32_f16(false, a, false, b, (short)0, c,
                                                false, false);
}

// ---- CDNA5 async copy: 16 bytes global -> LDS per lane (ASYNCcnt) ---------
__device__ __forceinline__ void async_ld16(unsigned lds_off, const void* g) {
  asm volatile("global_load_async_to_lds_b128 %0, %1, off"
               :: "v"(lds_off), "v"((unsigned long long)(uintptr_t)g)
               : "memory");
}
__device__ __forceinline__ void wait_async_le1() {
  asm volatile("s_wait_asynccnt 0x1" ::: "memory");
}
__device__ __forceinline__ void wait_async_le0() {
  asm volatile("s_wait_asynccnt 0x0" ::: "memory");
}
// Low 32 bits of a flat LDS address == LDS byte offset on AMDGPU.
__device__ __forceinline__ unsigned lds_off32(const void* p) {
  return (unsigned)(uintptr_t)p;
}

// A fragment (16x32 f16) from an f16 LDS tile, row pitch `pitch` halves.
// Contiguous K order (same as frag_b_wt): lanes 0-15 -> M=0..15; lane>>4 selects K half; VGPR v holds
// K = (lane>>4)*16 + v*2.
__device__ __forceinline__ v16h frag_a_lds(const half_t* tile, int pitch,
                                           int kbase, int lane) {
  const int m  = lane & 15;
  const int kh = lane >> 4;
  const half_t* row = tile + m * pitch + kbase;
  v16h a;
#pragma unroll
  for (int v = 0; v < 8; ++v) {
    const int k = (kh << 4) + (v << 1);
    a[2 * v]     = row[k];
    a[2 * v + 1] = row[k + 1];
  }
  return a;
}

// Same fragment but from an f32 LDS tile (async-copied raw), convert to f16.
__device__ __forceinline__ v16h frag_a_lds_f32(const float* tile, int pitch,
                                               int kbase, int lane) {
  const int m  = lane & 15;
  const int kh = lane >> 4;
  const float* row = tile + m * pitch + kbase;
  v16h a;
#pragma unroll
  for (int v = 0; v < 8; ++v) {
    const int k = (kh << 4) + (v << 1);
    a[2 * v]     = (half_t)row[k];
    a[2 * v + 1] = (half_t)row[k + 1];
  }
  return a;
}

// B fragment (32x16 f16) from a TRANSPOSED weight matrix WT[N][Ktot] (f16).
// Lanes 0-15 -> N (lanes 16-31 repeat N with K+16); VGPR v holds K=2v pair.
__device__ __forceinline__ v16h frag_b_wt(const half_t* WT, int Ktot, int nbase,
                                          int kbase, int lane) {
  const int n  = nbase + (lane & 15);
  const int kk = kbase + ((lane >> 4) << 4);
  const half_t* col = WT + (size_t)n * Ktot + kk;
  v16h b;
#pragma unroll
  for (int v = 0; v < 8; ++v) {
    b[2 * v]     = col[2 * v];
    b[2 * v + 1] = col[2 * v + 1];
  }
  return b;
}

// ---- small utility kernels -------------------------------------------------
__global__ void k_zero(float* p, int n) {
  int i = blockIdx.x * blockDim.x + threadIdx.x;
  if (i < n) p[i] = 0.0f;
}

// W[K][Nn] f32  ->  WT[Nn][K] f16
__global__ void k_cvt(const float* __restrict__ W, half_t* __restrict__ WT,
                      int K, int Nn) {
  int t = blockIdx.x * blockDim.x + threadIdx.x;
  if (t >= K * Nn) return;
  int k = t / Nn, n = t - k * Nn;
  WT[(size_t)n * K + k] = (half_t)W[(size_t)k * Nn + n];
}

// ---- edge aggregation: aggr_sum[dst] += x[src]; deg[dst] += 1 --------------
__global__ void k_edge(const float* __restrict__ x, const int* __restrict__ ei,
                       float* __restrict__ aggr_sum, float* __restrict__ deg) {
  int t = blockIdx.x * blockDim.x + threadIdx.x;
  if (t >= NUM_EDGES * D_IN) return;
  int e = t >> 6, f = t & 63;
  int src = ei[e];
  int dst = ei[NUM_EDGES + e];
  atomicAdd(&aggr_sum[dst * D_IN + f], x[src * D_IN + f]);
  if (f == 0) atomicAdd(&deg[dst], 1.0f);
}

// ---- per-graph node counts -------------------------------------------------
__global__ void k_pcnt(const int* __restrict__ batch, float* __restrict__ pcnt) {
  int i = blockIdx.x * blockDim.x + threadIdx.x;
  if (i < NUM_NODES) atomicAdd(&pcnt[batch[i]], 1.0f);
}

// ---- node GEMM: relu(aggr@Wl + bl + x@Wr), scattered into pooled sums ------
// NUM_NODES == 3125*16 exactly -> every tile is full, no guards needed.
__global__ void __launch_bounds__(256)
k_node(const float* __restrict__ aggr_sum, const float* __restrict__ deg,
       const float* __restrict__ x, const half_t* __restrict__ WlT,
       const half_t* __restrict__ WrT, const float* __restrict__ bl,
       const int* __restrict__ batch, float* __restrict__ pooled_sum) {
  __shared__ half_t Aag[16][72];
  __shared__ half_t Ax[16][72];
  __shared__ int    sh_batch[16];
  const int node0 = blockIdx.x * 16;

  if (threadIdx.x < 16) sh_batch[threadIdx.x] = batch[node0 + threadIdx.x];
#pragma unroll
  for (int it = 0; it < 4; ++it) {
    int i = threadIdx.x + it * 256;
    int r = i >> 6, c = i & 63;
    int node = node0 + r;
    float inv = 1.0f / fmaxf(deg[node], 1.0f);
    Aag[r][c] = (half_t)(aggr_sum[node * D_IN + c] * inv);
    Ax[r][c]  = (half_t)x[node * D_IN + c];
  }
  __syncthreads();

  const int wave = threadIdx.x >> 5;
  const int lane = threadIdx.x & 31;
#pragma unroll
  for (int t = 0; t < 2; ++t) {
    const int nbase = (wave + t * 8) * 16;
    v8f acc = {};
#pragma unroll
    for (int ks = 0; ks < 2; ++ks) {
      v16h aa = frag_a_lds(&Aag[0][0], 72, ks * 32, lane);
      v16h bL = frag_b_wt(WlT, D_IN, nbase, ks * 32, lane);
      acc = wmma_f16(aa, bL, acc);
      v16h ax = frag_a_lds(&Ax[0][0], 72, ks * 32, lane);
      v16h bR = frag_b_wt(WrT, D_IN, nbase, ks * 32, lane);
      acc = wmma_f16(ax, bR, acc);
    }
    const int n     = nbase + (lane & 15);
    const float bb  = bl[n];
    const int mbase = (lane >> 4) << 3;
#pragma unroll
    for (int i = 0; i < 8; ++i) {
      float val = fmaxf(acc[i] + bb, 0.0f);
      atomicAdd(&pooled_sum[sh_batch[mbase + i] * HID + n], val);
    }
  }
}

// ---- protein GEMM: protein_x = P @ Wp + bp ---------------------------------
// K=1280 in 20 chunks of 64; double-buffered async global->LDS staging.
__global__ void __launch_bounds__(256)
k_prot(const float* __restrict__ P, const half_t* __restrict__ WpT,
       const float* __restrict__ bp, float* __restrict__ protein_x) {
  __shared__ float Abuf[2][16][68];   // f32 chunk, row pitch 68 (bank stagger)
  const int g0   = blockIdx.x * 16;
  const int wave = threadIdx.x >> 5;
  const int lane = threadIdx.x & 31;

  // Each thread owns one 16-byte slice per chunk: row r, floats [c4, c4+4).
  const int r  = threadIdx.x >> 4;
  const int c4 = (threadIdx.x & 15) << 2;
  const float* gsrc = P + (size_t)(g0 + r) * D_PROT + c4;
  const unsigned ldst0 = lds_off32(&Abuf[0][r][c4]);
  const unsigned ldst1 = lds_off32(&Abuf[1][r][c4]);

  async_ld16(ldst0, gsrc);           // chunk 0 -> buf 0
  async_ld16(ldst1, gsrc + 64);      // chunk 1 -> buf 1

  v8f acc[2] = {{}, {}};
  for (int kc = 0; kc < D_PROT; kc += 64) {
    const int ibuf = (kc >> 6) & 1;
    if (kc + 64 < D_PROT) wait_async_le1();  // current chunk done (in-order)
    else                  wait_async_le0();
    __syncthreads();
#pragma unroll
    for (int t = 0; t < 2; ++t) {
      const int nbase = (wave + t * 8) * 16;
#pragma unroll
      for (int ks = 0; ks < 2; ++ks) {
        v16h a = frag_a_lds_f32(&Abuf[ibuf][0][0], 68, ks * 32, lane);
        v16h b = frag_b_wt(WpT, D_PROT, nbase, kc + ks * 32, lane);
        acc[t] = wmma_f16(a, b, acc[t]);
      }
    }
    __syncthreads();                 // all waves done reading this buffer
    if (kc + 128 < D_PROT)
      async_ld16(ibuf ? ldst1 : ldst0, gsrc + kc + 128);  // refill it
  }

#pragma unroll
  for (int t = 0; t < 2; ++t) {
    const int nbase = (wave + t * 8) * 16;
    const int n     = nbase + (lane & 15);
    const float bb  = bp[n];
#pragma unroll
    for (int i = 0; i < 8; ++i) {
      int m = i + ((lane >> 4) << 3);
      protein_x[(size_t)(g0 + m) * HID + n] = acc[t][i] + bb;
    }
  }
}

// ---- interaction GEMM: relu([pool | protein_x] @ Wi + bi) ------------------
__global__ void __launch_bounds__(256)
k_inter(const float* __restrict__ pooled_sum, const float* __restrict__ pcnt,
        const float* __restrict__ protein_x, const half_t* __restrict__ WiT,
        const float* __restrict__ bi, float* __restrict__ interaction) {
  __shared__ half_t At[16][72];
  const int g0   = blockIdx.x * 16;
  const int wave = threadIdx.x >> 5;
  const int lane = threadIdx.x & 31;
  v8f acc[2] = {{}, {}};

  for (int kc = 0; kc < 2 * HID; kc += 64) {
    __syncthreads();
#pragma unroll
    for (int it = 0; it < 4; ++it) {
      int i = threadIdx.x + it * 256;
      int r = i >> 6, c = i & 63;
      int g = g0 + r;
      int k = kc + c;
      float v;
      if (k < HID) {
        v = pooled_sum[g * HID + k] / fmaxf(pcnt[g], 1.0f);  // mean pool
      } else {
        v = protein_x[g * HID + (k - HID)];                  // concat
      }
      At[r][c] = (half_t)v;
    }
    __syncthreads();
#pragma unroll
    for (int t = 0; t < 2; ++t) {
      const int nbase = (wave + t * 8) * 16;
#pragma unroll
      for (int ks = 0; ks < 2; ++ks) {
        v16h a = frag_a_lds(&At[0][0], 72, ks * 32, lane);
        v16h b = frag_b_wt(WiT, 2 * HID, nbase, kc + ks * 32, lane);
        acc[t] = wmma_f16(a, b, acc[t]);
      }
    }
  }
#pragma unroll
  for (int t = 0; t < 2; ++t) {
    const int nbase = (wave + t * 8) * 16;
    const int n     = nbase + (lane & 15);
    const float bb  = bi[n];
#pragma unroll
    for (int i = 0; i < 8; ++i) {
      int m = i + ((lane >> 4) << 3);
      interaction[(size_t)(g0 + m) * HID + n] = fmaxf(acc[t][i] + bb, 0.0f);
    }
  }
}

// ---- output head: out[g] = interaction[g] . Wo + bo  (wave per graph) ------
__global__ void __launch_bounds__(256)
k_out(const float* __restrict__ interaction, const float* __restrict__ Wo,
      const float* __restrict__ bo, float* __restrict__ out) {
  const int g    = blockIdx.x * 8 + (threadIdx.x >> 5);
  const int lane = threadIdx.x & 31;
  if (g >= NUM_GRAPHS) return;
  float s = 0.0f;
#pragma unroll
  for (int k = lane; k < HID; k += 32) s += interaction[g * HID + k] * Wo[k];
#pragma unroll
  for (int off = 16; off > 0; off >>= 1) s += __shfl_xor(s, off, 32);
  if (lane == 0) out[g] = s + bo[0];
}

// ---------------------------------------------------------------------------
extern "C" void kernel_launch(void* const* d_in, const int* in_sizes, int n_in,
                              void* d_out, int out_size, void* d_ws,
                              size_t ws_size, hipStream_t stream) {
  (void)in_sizes; (void)n_in; (void)out_size; (void)ws_size;
  const float* x    = (const float*)d_in[0];
  const float* prot = (const float*)d_in[1];
  const float* Wl   = (const float*)d_in[2];
  const float* bl   = (const float*)d_in[3];
  const float* Wr   = (const float*)d_in[4];
  const float* Wp   = (const float*)d_in[5];
  const float* bp   = (const float*)d_in[6];
  const float* Wi   = (const float*)d_in[7];
  const float* bi   = (const float*)d_in[8];
  const float* Wo   = (const float*)d_in[9];
  const float* bo   = (const float*)d_in[10];
  const int*   ei   = (const int*)d_in[11];
  const int*   bat  = (const int*)d_in[12];
  float* out = (float*)d_out;

  // Workspace layout (floats, then halves). Total ~17.2 MB.
  float* ws          = (float*)d_ws;
  float* aggr_sum    = ws;                              // 50000*64
  float* deg         = aggr_sum + NUM_NODES * D_IN;     // 50000
  float* pooled_sum  = deg + NUM_NODES;                 // 1024*256
  float* pooled_cnt  = pooled_sum + NUM_GRAPHS * HID;   // 1024
  float* protein_x   = pooled_cnt + NUM_GRAPHS;         // 1024*256
  float* interaction = protein_x + NUM_GRAPHS * HID;    // 1024*256
  half_t* WlT = (half_t*)(interaction + NUM_GRAPHS * HID);
  half_t* WrT = WlT + HID * D_IN;
  half_t* WpT = WrT + HID * D_IN;
  half_t* WiT = WpT + HID * D_PROT;

  const int zero_n = NUM_NODES * D_IN + NUM_NODES + NUM_GRAPHS * HID + NUM_GRAPHS;
  k_zero<<<(zero_n + 255) / 256, 256, 0, stream>>>(aggr_sum, zero_n);

  k_cvt<<<(D_IN * HID + 255) / 256, 256, 0, stream>>>(Wl, WlT, D_IN, HID);
  k_cvt<<<(D_IN * HID + 255) / 256, 256, 0, stream>>>(Wr, WrT, D_IN, HID);
  k_cvt<<<(D_PROT * HID + 255) / 256, 256, 0, stream>>>(Wp, WpT, D_PROT, HID);
  k_cvt<<<(2 * HID * HID + 255) / 256, 256, 0, stream>>>(Wi, WiT, 2 * HID, HID);

  k_edge<<<(NUM_EDGES * D_IN + 255) / 256, 256, 0, stream>>>(x, ei, aggr_sum, deg);
  k_pcnt<<<(NUM_NODES + 255) / 256, 256, 0, stream>>>(bat, pooled_cnt);

  k_node<<<NUM_NODES / 16, 256, 0, stream>>>(aggr_sum, deg, x, WlT, WrT,
                                             bl, bat, pooled_sum);
  k_prot<<<NUM_GRAPHS / 16, 256, 0, stream>>>(prot, WpT, bp, protein_x);
  k_inter<<<NUM_GRAPHS / 16, 256, 0, stream>>>(pooled_sum, pooled_cnt,
                                               protein_x, WiT, bi, interaction);
  k_out<<<NUM_GRAPHS / 8, 256, 0, stream>>>(interaction, Wo, bo, out);
}
